// ImageLinearAttention_40999757807965
// MI455X (gfx1250) — matrix operand order repaired and checked by reading, not compiled, not run on hardware
//
#include <hip/hip_runtime.h>
#include <cstdint>

// ---------------------------------------------------------------------------
// ImageLinearAttention fused pipeline for gfx1250 (wave32, WMMA 16x16x32 f16)
// ---------------------------------------------------------------------------

typedef _Float16 f16;
typedef _Float16 f16x8_t  __attribute__((ext_vector_type(8)));
typedef _Float16 f16x16_t __attribute__((ext_vector_type(16)));
typedef float    f32x8_t  __attribute__((ext_vector_type(8)));
typedef int      v4i_vs   __attribute__((vector_size(16)));   // int4, matches builtin proto

#define BATCH 16
#define CHAN  256
#define NPOS  4096
#define HEADS 8
#define KD    64
#define QD    (HEADS * KD)   // 512
#define KVSPLIT 16           // n-splits for the kv/context kernel

#define AS1 __attribute__((address_space(1)))
#define AS3 __attribute__((address_space(3)))

#if defined(__has_builtin)
#if __has_builtin(__builtin_amdgcn_global_load_async_to_lds_b128) && \
    __has_builtin(__builtin_amdgcn_s_wait_asynccnt)
#define HAVE_ASYNC_LDS 1
#endif
#endif
#ifndef HAVE_ASYNC_LDS
#define HAVE_ASYNC_LDS 0
#endif

// D = A(16x32) * B(32x16) + C, f32 accumulate
__device__ __forceinline__ f32x8_t wmma16(f16x16_t a, f16x16_t b, f32x8_t c) {
  return __builtin_amdgcn_wmma_f32_16x16x32_f16(
      /*neg_a=*/false, a, /*neg_b=*/false, b,
      /*c_mod=*/(short)0, c, /*reuse_a=*/false, /*reuse_b=*/false);
}

// A-matrix fragment (16x32, row-major source, row stride ld halfwords).
// Lanes 0-15 -> M=lane, K = {0..7, 16..23}; lanes 16-31 -> K += 8.
__device__ __forceinline__ f16x16_t load_a_frag(const f16* p0, int ld, int lane) {
  const f16* p = p0 + (lane & 15) * ld + ((lane >> 4) << 3);
  f16x8_t lo = *(const f16x8_t*)(p);
  f16x8_t hi = *(const f16x8_t*)(p + 16);
  return __builtin_shufflevector(lo, hi, 0,1,2,3,4,5,6,7,8,9,10,11,12,13,14,15);
}

// B-matrix fragment (32x16) from K-contiguous storage M[n][k], row stride ld.
// Lanes 0-15 hold K = {0..7, 16..23} for column N=lane; lanes 16-31 -> K += 8.
__device__ __forceinline__ f16x16_t load_b_frag(const f16* p0, int ld, int lane) {
  const f16* p = p0 + (lane & 15) * ld + ((lane >> 4) << 3);
  f16x8_t lo = *(const f16x8_t*)(p);
  f16x8_t hi = *(const f16x8_t*)(p + 16);
  return __builtin_shufflevector(lo, hi, 0,1,2,3,4,5,6,7,8,9,10,11,12,13,14,15);
}

// Vectorized X-tile fill: Xh is n-major [b][n][c]; copy 64 rows x 256 halves.
// Uses the CDNA5 async global->LDS path when the toolchain exposes it.
__device__ __forceinline__ void fill_x_tile(f16 Xs[64][CHAN + 8],
                                            const f16* __restrict__ xrow,
                                            int tid) {
#if HAVE_ASYNC_LDS
#pragma unroll
  for (int i = tid; i < 64 * (CHAN / 8); i += 256) {
    const int j = i >> 5;
    const int cc = (i & 31) << 3;
    const f16* g = &xrow[(size_t)j * CHAN + cc];
    f16* l = &Xs[j][cc];
    __builtin_amdgcn_global_load_async_to_lds_b128(
        (AS1 v4i_vs*)(uintptr_t)g,
        (AS3 v4i_vs*)(unsigned)(uintptr_t)l,
        /*offset=*/0, /*cpol=*/0);
  }
  __builtin_amdgcn_s_wait_asynccnt(0);
#else
#pragma unroll
  for (int i = tid; i < 64 * (CHAN / 8); i += 256) {
    const int j = i >> 5;
    const int cc = (i & 31) << 3;
    *(f16x8_t*)&Xs[j][cc] = *(const f16x8_t*)&xrow[(size_t)j * CHAN + cc];
  }
#endif
}

// ---------------------------------------------------------------------------
// Staging: weights f32 -> f16 (layout preserved)
// ---------------------------------------------------------------------------
__global__ __launch_bounds__(256) void cvt_f32_f16(const float* __restrict__ in,
                                                   f16* __restrict__ out, int n) {
  int i = blockIdx.x * 256 + threadIdx.x;
  if (i < n) out[i] = (f16)in[i];
}

// Staging: x [b][c][n] f32 -> Xh [b][n][c] f16, LDS-tiled transpose.
// grid: (n/64, CHAN/64, BATCH)
__global__ __launch_bounds__(256) void cvt_x_transpose(const float* __restrict__ x,
                                                       f16* __restrict__ Xh) {
  __shared__ f16 T[64][64 + 8];
  const int tid = threadIdx.x;
  const int n0 = blockIdx.x * 64, c0 = blockIdx.y * 64, b = blockIdx.z;
  const float* xb = x + ((size_t)b * CHAN + c0) * NPOS + n0;
#pragma unroll
  for (int i = tid; i < 64 * 64; i += 256) {          // coalesced along n
    const int ci = i >> 6, nj = i & 63;
    T[ci][nj] = (f16)xb[(size_t)ci * NPOS + nj];
  }
  __syncthreads();
  f16* ob = Xh + ((size_t)b * NPOS + n0) * CHAN + c0;
#pragma unroll
  for (int i = tid; i < 64 * 8; i += 256) {           // coalesced along c
    const int nj = i >> 3, cc = (i & 7) << 3;
    f16x8_t v;
#pragma unroll
    for (int r = 0; r < 8; ++r) v[r] = T[cc + r][nj];
    *(f16x8_t*)&ob[(size_t)nj * CHAN + cc] = v;
  }
}

// ---------------------------------------------------------------------------
// Zero-init for global accumulators (ws is poisoned; must re-zero every call)
// ---------------------------------------------------------------------------
__global__ __launch_bounds__(256) void zero_f32(float* __restrict__ p, int n) {
  int i = blockIdx.x * 256 + threadIdx.x;
  if (i < n) p[i] = 0.0f;
}

// ---------------------------------------------------------------------------
// Kernel 1: partial K/V projection + exp + partial ksum / context.
// grid (KVSPLIT, BATCH); each block handles 256 positions (4 x 64-col tiles).
// Unnormalized context and ksum accumulated with global f32 atomics.
// ---------------------------------------------------------------------------
__global__ __launch_bounds__(256) void attn_kv_partial(
    const f16* __restrict__ Xh, const f16* __restrict__ Wk16,
    const float* __restrict__ bk, const f16* __restrict__ Wv16,
    const float* __restrict__ bv, float* __restrict__ ctxAcc,
    float* __restrict__ ksumAcc) {
  __shared__ f16 Xs[64][CHAN + 8];  // X tile, n-major rows (B layout)
  __shared__ f16 Ek[64][64 + 8];    // exp(K) tile, d-major (A layout)
  __shared__ f16 Vs[64][64 + 8];    // V tile, e-major (B layout, K = n)
  __shared__ float ksum[64];

  const int tid = threadIdx.x;
  const int lane = tid & 31;
  const int wave = tid >> 5;
  const int b = blockIdx.y;
  const int nbase = blockIdx.x * (NPOS / KVSPLIT);   // 256 positions per block
  const f16* xb = Xh + (size_t)b * NPOS * CHAN;

  if (tid < 64) ksum[tid] = 0.0f;

  const f32x8_t vzero = {0.f,0.f,0.f,0.f,0.f,0.f,0.f,0.f};
  f32x8_t cacc[2] = {vzero, vzero};  // 2 context 16x16 tiles per wave

  for (int t = 0; t < (NPOS / KVSPLIT) / 64; ++t) {
    const int n0 = nbase + t * 64;
    __syncthreads();  // prior iteration consumers of Xs/Ek/Vs are done
    fill_x_tile(Xs, xb + (size_t)n0 * CHAN, tid);
    if (t < 3 && tid < 64)
      __builtin_prefetch(&xb[(size_t)(n0 + 64 + tid) * CHAN], 0, 1);
    __syncthreads();

    {
      const bool isK = (wave < 4);
      const int mrow = (wave & 3) * 16;
      const f16* wbase = (isK ? Wk16 : Wv16) + (size_t)mrow * CHAN;
      f32x8_t acc[4] = {vzero, vzero, vzero, vzero};
#pragma unroll
      for (int k0 = 0; k0 < CHAN; k0 += 32) {
        f16x16_t afrag = load_a_frag(wbase + k0, CHAN, lane);
#pragma unroll
        for (int nt = 0; nt < 4; ++nt) {
          f16x16_t bfrag = load_b_frag(&Xs[nt * 16][k0], CHAN + 8, lane);
          acc[nt] = wmma16(afrag, bfrag, acc[nt]);
        }
      }
      const int rhi = (lane >> 4) << 3;
      const int ncol = lane & 15;
      if (isK) {
#pragma unroll
        for (int nt = 0; nt < 4; ++nt)
#pragma unroll
          for (int r = 0; r < 8; ++r) {
            const int m = mrow + rhi + r;
            const float e = __expf(acc[nt][r] + bk[m]);
            Ek[m][nt * 16 + ncol] = (f16)e;
            atomicAdd(&ksum[m], e);  // ds_add_f32
          }
      } else {
#pragma unroll
        for (int nt = 0; nt < 4; ++nt)
#pragma unroll
          for (int r = 0; r < 8; ++r) {
            const int m = mrow + rhi + r;
            Vs[m][nt * 16 + ncol] = (f16)(acc[nt][r] + bv[m]);
          }
      }
    }
    __syncthreads();

    // context += expK(64x64) * V^T(64x64); 16 tiles over 8 waves
#pragma unroll
    for (int i = 0; i < 2; ++i) {
      const int idx = wave * 2 + i, mt = idx >> 2, nt = idx & 3;
#pragma unroll
      for (int k0 = 0; k0 < 64; k0 += 32) {
        f16x16_t afrag = load_a_frag(&Ek[mt * 16][k0], 64 + 8, lane);
        f16x16_t bfrag = load_b_frag(&Vs[nt * 16][k0], 64 + 8, lane);
        cacc[i] = wmma16(afrag, bfrag, cacc[i]);
      }
    }
  }
  __syncthreads();
  {
    const int rhi = (lane >> 4) << 3;
    const int ncol = lane & 15;
    float* cb = ctxAcc + (size_t)b * (KD * KD);
#pragma unroll
    for (int i = 0; i < 2; ++i) {
      const int idx = wave * 2 + i, mt = idx >> 2, nt = idx & 3;
#pragma unroll
      for (int r = 0; r < 8; ++r) {
        const int d = mt * 16 + rhi + r;
        const int e = nt * 16 + ncol;
        atomicAdd(&cb[d * 64 + e], cacc[i][r]);     // global_atomic_add_f32
      }
    }
    if (tid < 64) atomicAdd(&ksumAcc[b * 64 + tid], ksum[tid]);
  }
}

// ctx[b][d][e] = ctxAcc / ksumAcc[b][d]
__global__ __launch_bounds__(256) void ctx_finalize(const float* __restrict__ ctxAcc,
                                                    const float* __restrict__ ksumAcc,
                                                    float* __restrict__ ctx) {
  const int i = blockIdx.x * 256 + threadIdx.x;
  if (i < BATCH * KD * KD) {
    const int b = i >> 12, d = (i >> 6) & 63;
    ctx[i] = ctxAcc[i] / ksumAcc[b * 64 + d];
  }
}

// ---------------------------------------------------------------------------
// Kernel 2: fused Q-proj -> softmax(d) -> per-head ctx mul -> Wo proj.
// One block per (batch, 64-column tile). 8 waves.
// ---------------------------------------------------------------------------
__global__ __launch_bounds__(256) void attn_main(
    const f16* __restrict__ Xh, const f16* __restrict__ Wq16,
    const float* __restrict__ bq, const f16* __restrict__ Wo16,
    const float* __restrict__ bo, const float* __restrict__ ctx,
    float* __restrict__ out) {
  __shared__ f16 Xs[64][CHAN + 8];   // X tile, n-major rows
  __shared__ f16 Qs[64][QD + 8];     // Q (then O) transposed [n][h*64+d]
  __shared__ f16 Ct[64][64 + 8];     // ctx^T: Ct[e][d]

  const int tid = threadIdx.x;
  const int lane = tid & 31;
  const int wave = tid >> 5;
  const int b = blockIdx.x >> 6;
  const int n0 = (blockIdx.x & 63) * 64;
  const int rhi = (lane >> 4) << 3;
  const int ncol = lane & 15;
  const f32x8_t vzero = {0.f,0.f,0.f,0.f,0.f,0.f,0.f,0.f};

  fill_x_tile(Xs, Xh + ((size_t)b * NPOS + n0) * CHAN, tid);
  const float* cb = ctx + (size_t)b * (KD * KD);
#pragma unroll
  for (int i = tid; i < KD * KD; i += 256) {
    const int d = i >> 6, e = i & 63;
    Ct[e][d] = (f16)cb[i];
  }
  __syncthreads();

  // ---- Q GEMM: rows [64*wave, 64*wave+64) of Wq(512x256) @ Xtile ----
  {
    f32x8_t acc[4][4];
#pragma unroll
    for (int mt = 0; mt < 4; ++mt)
#pragma unroll
      for (int nt = 0; nt < 4; ++nt) acc[mt][nt] = vzero;

    const f16* wbase = Wq16 + (size_t)(wave * 64) * CHAN;
#pragma unroll
    for (int k0 = 0; k0 < CHAN; k0 += 32) {
      f16x16_t bfrag[4];
#pragma unroll
      for (int nt = 0; nt < 4; ++nt)
        bfrag[nt] = load_b_frag(&Xs[nt * 16][k0], CHAN + 8, lane);
#pragma unroll
      for (int mt = 0; mt < 4; ++mt) {
        f16x16_t afrag = load_a_frag(wbase + (size_t)mt * 16 * CHAN + k0, CHAN, lane);
#pragma unroll
        for (int nt = 0; nt < 4; ++nt)
          acc[mt][nt] = wmma16(afrag, bfrag[nt], acc[mt][nt]);
      }
    }
#pragma unroll
    for (int mt = 0; mt < 4; ++mt) {
      const int mbase = wave * 64 + mt * 16 + rhi;
#pragma unroll
      for (int nt = 0; nt < 4; ++nt) {
        const int n = nt * 16 + ncol;
        f16x8_t v;
#pragma unroll
        for (int r = 0; r < 8; ++r)
          v[r] = (f16)(acc[mt][nt][r] + bq[mbase + r]);
        *(f16x8_t*)&Qs[n][mbase] = v;
      }
    }
  }
  __syncthreads();

  // ---- softmax over d for each (n, head): contiguous 64 halves ----
  for (int i = tid; i < 64 * HEADS; i += 256) {
    const int n = i >> 3, h = i & 7;
    f16* p = &Qs[n][h * 64];
    float mx = -1e30f;
#pragma unroll 8
    for (int d = 0; d < 64; ++d) {
      const float v = (float)p[d];
      mx = v > mx ? v : mx;
    }
    float s = 0.0f;
#pragma unroll 8
    for (int d = 0; d < 64; ++d) s += __expf((float)p[d] - mx);
    const float inv = 1.0f / s;
#pragma unroll 8
    for (int d = 0; d < 64; ++d)
      p[d] = (f16)(__expf((float)p[d] - mx) * inv);
  }
  __syncthreads();

  // ---- per-head O = ctx^T(64x64) @ Qs_h(64x64); wave w owns head w ----
  {
    const int h = wave;
    f32x8_t acc[4][4];
#pragma unroll
    for (int mt = 0; mt < 4; ++mt)
#pragma unroll
      for (int nt = 0; nt < 4; ++nt) acc[mt][nt] = vzero;
#pragma unroll
    for (int k0 = 0; k0 < 64; k0 += 32) {
      f16x16_t bfrag[4];
#pragma unroll
      for (int nt = 0; nt < 4; ++nt)
        bfrag[nt] = load_b_frag(&Qs[nt * 16][h * 64 + k0], QD + 8, lane);
#pragma unroll
      for (int mt = 0; mt < 4; ++mt) {
        f16x16_t afrag = load_a_frag(&Ct[mt * 16][k0], 64 + 8, lane);
#pragma unroll
        for (int nt = 0; nt < 4; ++nt)
          acc[mt][nt] = wmma16(afrag, bfrag[nt], acc[mt][nt]);
      }
    }
    // overwrite this head's slice of Qs with O (wave-private columns)
#pragma unroll
    for (int mt = 0; mt < 4; ++mt) {
      const int ebase = mt * 16 + rhi;
#pragma unroll
      for (int nt = 0; nt < 4; ++nt) {
        const int n = nt * 16 + ncol;
        f16x8_t v;
#pragma unroll
        for (int r = 0; r < 8; ++r) v[r] = (f16)acc[mt][nt][r];
        *(f16x8_t*)&Qs[n][h * 64 + ebase] = v;
      }
    }
  }
  __syncthreads();

  // ---- Y = Wo(256x512) @ O(512x64) + bo; wave w owns rows [32w, 32w+32) ----
  {
    f32x8_t acc[2][4];
#pragma unroll
    for (int mt = 0; mt < 2; ++mt)
#pragma unroll
      for (int nt = 0; nt < 4; ++nt) acc[mt][nt] = vzero;

    const f16* wbase = Wo16 + (size_t)(wave * 32) * QD;
#pragma unroll
    for (int k0 = 0; k0 < QD; k0 += 32) {
      f16x16_t bfrag[4];
#pragma unroll
      for (int nt = 0; nt < 4; ++nt)
        bfrag[nt] = load_b_frag(&Qs[nt * 16][k0], QD + 8, lane);
#pragma unroll
      for (int mt = 0; mt < 2; ++mt) {
        f16x16_t afrag = load_a_frag(wbase + (size_t)mt * 16 * QD + k0, QD, lane);
#pragma unroll
        for (int nt = 0; nt < 4; ++nt)
          acc[mt][nt] = wmma16(afrag, bfrag[nt], acc[mt][nt]);
      }
    }
    float* ob = out + (size_t)b * CHAN * NPOS;
#pragma unroll
    for (int mt = 0; mt < 2; ++mt) {
      const int mbase = wave * 32 + mt * 16 + rhi;
#pragma unroll
      for (int nt = 0; nt < 4; ++nt) {
        const int col = n0 + nt * 16 + ncol;
#pragma unroll
        for (int r = 0; r < 8; ++r) {
          const int m = mbase + r;
          ob[(size_t)m * NPOS + col] = acc[mt][nt][r] + bo[m];
        }
      }
    }
  }
}

// ---------------------------------------------------------------------------
// Host launcher
// ---------------------------------------------------------------------------
extern "C" void kernel_launch(void* const* d_in, const int* in_sizes, int n_in,
                              void* d_out, int out_size, void* d_ws, size_t ws_size,
                              hipStream_t stream) {
  (void)in_sizes; (void)n_in; (void)out_size; (void)ws_size;
  const float* x  = (const float*)d_in[0];
  const float* Wq = (const float*)d_in[1];
  const float* bq = (const float*)d_in[2];
  const float* Wk = (const float*)d_in[3];
  const float* bk = (const float*)d_in[4];
  const float* Wv = (const float*)d_in[5];
  const float* bv = (const float*)d_in[6];
  const float* Wo = (const float*)d_in[7];
  const float* bo = (const float*)d_in[8];
  float* out = (float*)d_out;

  const int XN  = BATCH * CHAN * NPOS;   // 16,777,216
  const int WQN = QD * CHAN;             // 131,072
  const int WKN = KD * CHAN;             // 16,384
  const int WVN = KD * CHAN;             // 16,384
  const int WON = CHAN * QD;             // 131,072
  const int CTXN = BATCH * KD * KD;      // 65,536
  const int KSN  = BATCH * KD;           // 1,024

  f16* Xh   = (f16*)d_ws;                // n-major: [b][n][c]
  f16* Wq16 = Xh + XN;
  f16* Wk16 = Wq16 + WQN;
  f16* Wv16 = Wk16 + WKN;
  f16* Wo16 = Wv16 + WVN;
  float* ctx     = (float*)(Wo16 + WON);
  float* ctxAcc  = ctx + CTXN;
  float* ksumAcc = ctxAcc + CTXN;        // contiguous with ctxAcc

  cvt_x_transpose<<<dim3(NPOS / 64, CHAN / 64, BATCH), 256, 0, stream>>>(x, Xh);
  cvt_f32_f16<<<(WQN + 255) / 256, 256, 0, stream>>>(Wq, Wq16, WQN);
  cvt_f32_f16<<<(WKN + 255) / 256, 256, 0, stream>>>(Wk, Wk16, WKN);
  cvt_f32_f16<<<(WVN + 255) / 256, 256, 0, stream>>>(Wv, Wv16, WVN);
  cvt_f32_f16<<<(WON + 255) / 256, 256, 0, stream>>>(Wo, Wo16, WON);
  zero_f32<<<(CTXN + KSN + 255) / 256, 256, 0, stream>>>(ctxAcc, CTXN + KSN);

  attn_kv_partial<<<dim3(KVSPLIT, BATCH), 256, 0, stream>>>(Xh, Wk16, bk, Wv16, bv,
                                                            ctxAcc, ksumAcc);
  ctx_finalize<<<(CTXN + 255) / 256, 256, 0, stream>>>(ctxAcc, ksumAcc, ctx);
  attn_main<<<BATCH * (NPOS / 64), 256, 0, stream>>>(Xh, Wq16, bq, Wo16, bo, ctx, out);
}
